// HybridLayer_83013127897575
// MI455X (gfx1250) — matrix operand order repaired and checked by reading, not compiled, not run on hardware
//
#include <hip/hip_runtime.h>

// ---------------------------------------------------------------------------
// HybridLayer: out = x @ W.T + bias, where W is synthesized per call:
//   W = sigmoid(alpha)*tanh(W_fourier)*std + sigmoid(beta)*tanh(lora_A@lora_B)*std
// Memory-bound on MI455X (~420MB moved vs 275 GFLOP) -> bf16 WMMA GEMM with
// L2-resident weights, LDS double buffering via gfx1250 async global->LDS
// copies (ASYNCcnt), fp32 accumulation.
// ---------------------------------------------------------------------------

typedef __attribute__((ext_vector_type(16))) __bf16 v16bf;
typedef __attribute__((ext_vector_type(8)))  __bf16 v8bf;
typedef __attribute__((ext_vector_type(4)))  __bf16 v4bf;
typedef __attribute__((ext_vector_type(8)))  float  v8f;
typedef __attribute__((ext_vector_type(8)))  unsigned int v8u;
typedef __attribute__((ext_vector_type(4)))  int i32x4;

#define IN_F   4096
#define OUT_F  4096
#define NFREQ  16
#define NTOK   8192

// gfx1250 async global->LDS copy path (ASYNCcnt). Builtin signature (from the
// compiler diagnostic): (int4 AS1*, int4 AS3*, imm offset, imm cpol).
#if defined(__gfx1250__) && \
    __has_builtin(__builtin_amdgcn_global_load_async_to_lds_b128) && \
    __has_builtin(__builtin_amdgcn_s_wait_asynccnt)
#define USE_ASYNC_LDS 1
#define GPTR(p) ((__attribute__((address_space(1))) i32x4*)(p))
#define LPTR(p) ((__attribute__((address_space(3))) i32x4*)(p))
#else
#define USE_ASYNC_LDS 0
#endif

// ------------------------------- Phase 1 -----------------------------------
// S[k*OUT_F+o] = amp[k]*sin(r_o*rf[k]+rp[k]);  C[k*IN_F+i] = cos(c_i*cf[k]+cp[k])
__global__ void k_fourier_sc(const float* __restrict__ amp, const float* __restrict__ rf,
                             const float* __restrict__ cf, const float* __restrict__ rp,
                             const float* __restrict__ cp,
                             float* __restrict__ S, float* __restrict__ C) {
  int idx = blockIdx.x * blockDim.x + threadIdx.x;   // [0, NFREQ*4096)
  int k = idx >> 12;
  int j = idx & 4095;
  const float step = 6.28318530717958647692f / 4095.0f;  // linspace(0, 2pi, 4096)
  float pos = step * (float)j;
  S[idx] = amp[k] * sinf(pos * rf[k] + rp[k]);
  C[idx] = cosf(pos * cf[k] + cp[k]);
}

// ------------------------------- Phase 2 -----------------------------------
// W[o,i] (bf16, row-major, i==K contiguous). Each thread makes 4 elements.
__global__ void k_wsynth(const float* __restrict__ S, const float* __restrict__ C,
                         const float* __restrict__ lA, const float* __restrict__ lB,
                         const float* __restrict__ alpha, const float* __restrict__ beta,
                         __bf16* __restrict__ Wb) {
  const int o  = blockIdx.y;
  const int i0 = (blockIdx.x * blockDim.x + threadIdx.x) * 4;

  float sk[NFREQ];
#pragma unroll
  for (int k = 0; k < NFREQ; ++k) sk[k] = S[k * OUT_F + o];   // uniform per block row

  const float a0 = lA[o * 2 + 0];
  const float a1 = lA[o * 2 + 1];
  const float4 lb0 = *(const float4*)&lB[i0];
  const float4 lb1 = *(const float4*)&lB[IN_F + i0];

  float f0 = 0.f, f1 = 0.f, f2 = 0.f, f3 = 0.f;
#pragma unroll
  for (int k = 0; k < NFREQ; ++k) {
    float4 c = *(const float4*)&C[k * IN_F + i0];
    float s = sk[k];
    f0 += s * c.x; f1 += s * c.y; f2 += s * c.z; f3 += s * c.w;
  }

  const float sa = 1.0f / (1.0f + expf(-alpha[0]));
  const float sb = 1.0f / (1.0f + expf(-beta[0]));
  const float stdv = 0.015625f;                 // sqrt(2/(4096+4096)) == 1/64 exactly
  const float as = sa * stdv, bs = sb * stdv;

  float w0 = as * tanhf(f0) + bs * tanhf(a0 * lb0.x + a1 * lb1.x);
  float w1 = as * tanhf(f1) + bs * tanhf(a0 * lb0.y + a1 * lb1.y);
  float w2 = as * tanhf(f2) + bs * tanhf(a0 * lb0.z + a1 * lb1.z);
  float w3 = as * tanhf(f3) + bs * tanhf(a0 * lb0.w + a1 * lb1.w);

  v4bf wv;
  wv[0] = (__bf16)w0; wv[1] = (__bf16)w1; wv[2] = (__bf16)w2; wv[3] = (__bf16)w3;
  *(v4bf*)&Wb[(size_t)o * IN_F + i0] = wv;
}

// ------------------------------- Phase 3 -----------------------------------
__global__ void k_xcvt(const float* __restrict__ x, __bf16* __restrict__ xb) {
  size_t idx = ((size_t)blockIdx.x * blockDim.x + threadIdx.x) * 8;
  float4 f0 = *(const float4*)&x[idx];
  float4 f1 = *(const float4*)&x[idx + 4];
  v8bf v;
  v[0] = (__bf16)f0.x; v[1] = (__bf16)f0.y; v[2] = (__bf16)f0.z; v[3] = (__bf16)f0.w;
  v[4] = (__bf16)f1.x; v[5] = (__bf16)f1.y; v[6] = (__bf16)f1.z; v[7] = (__bf16)f1.w;
  *(v8bf*)&xb[idx] = v;
}

// ------------------------------- Phase 4: GEMM -----------------------------
// out[t,o] = sum_i Xb[t,i]*Wb[o,i] + bias[o]   (NT: both operands K-contiguous)
#define BM   128
#define BN   128
#define BK   32
#define BKP  40      // padded row (80B stride: 16B aligned, conflict-free cols)
#define KDIM 4096

__global__ __launch_bounds__(256) void k_gemm(const __bf16* __restrict__ X,
                                              const __bf16* __restrict__ W,
                                              const float* __restrict__ bias,
                                              float* __restrict__ out) {
  __shared__ __bf16 As[2][BM][BKP];   // 10 KB each buffer
  __shared__ __bf16 Bs[2][BN][BKP];

  const int tid  = threadIdx.x;
  const int lane = tid & 31;
  const int wave = tid >> 5;          // 8 waves: 2(M) x 4(N), wave tile 64x32
  const int wm   = (wave & 1) * 64;
  const int wn   = (wave >> 1) * 32;
  const int bm   = blockIdx.y * BM;
  const int bn   = blockIdx.x * BN;
  const int l15  = lane & 15;
  const int lhi  = lane >> 4;

  // tile loaders: 256 threads x 32B each for A and B (8KB tiles)
  const int lrow = tid >> 1;          // 0..127
  const int lcol = (tid & 1) * 16;    // element offset 0 or 16

  const __bf16* gA = X + (size_t)(bm + lrow) * KDIM + lcol;
  const __bf16* gB = W + (size_t)(bn + lrow) * KDIM + lcol;

  v8f acc[4][2];
  v8f zero = 0.0f;
#pragma unroll
  for (int mi = 0; mi < 4; ++mi)
#pragma unroll
    for (int ni = 0; ni < 2; ++ni) acc[mi][ni] = zero;

  const int NK = KDIM / BK;  // 128

#if USE_ASYNC_LDS
  // -------- async global->LDS pipeline (ASYNCcnt-tracked, no VGPR staging) --
  auto issue_tile = [&](int buf, int kk) {
    const __bf16* ga = gA + kk;
    const __bf16* gb = gB + kk;
    __bf16* la = &As[buf][lrow][lcol];
    __bf16* lb = &Bs[buf][lrow][lcol];
    __builtin_amdgcn_global_load_async_to_lds_b128(GPTR(ga),     LPTR(la),     0, 0);
    __builtin_amdgcn_global_load_async_to_lds_b128(GPTR(ga + 8), LPTR(la + 8), 0, 0);
    __builtin_amdgcn_global_load_async_to_lds_b128(GPTR(gb),     LPTR(lb),     0, 0);
    __builtin_amdgcn_global_load_async_to_lds_b128(GPTR(gb + 8), LPTR(lb + 8), 0, 0);
  };
  issue_tile(0, 0);                       // prime buffer 0 (4 asyncs in flight)
#else
  uint4 ra0, ra1, rb0, rb1;
  {  // prime buffer 0 through registers
    const uint4* pa = (const uint4*)gA;  ra0 = pa[0]; ra1 = pa[1];
    const uint4* pb = (const uint4*)gB;  rb0 = pb[0]; rb1 = pb[1];
    *(uint4*)&As[0][lrow][lcol]     = ra0;
    *(uint4*)&As[0][lrow][lcol + 8] = ra1;
    *(uint4*)&Bs[0][lrow][lcol]     = rb0;
    *(uint4*)&Bs[0][lrow][lcol + 8] = rb1;
  }
  __syncthreads();
#endif

  for (int kt = 0; kt < NK; ++kt) {
    const int cur = kt & 1;
    const int nxt = cur ^ 1;

#if USE_ASYNC_LDS
    if (kt + 1 < NK) {
      issue_tile(nxt, (kt + 1) * BK);       // 8 asyncs in flight
      __builtin_amdgcn_s_wait_asynccnt(4);  // in-order: current tile's 4 done
    } else {
      __builtin_amdgcn_s_wait_asynccnt(0);
    }
    __syncthreads();                        // current tile visible to all waves
#else
    if (kt + 1 < NK) {  // prefetch next tile into registers
      const uint4* pa = (const uint4*)(gA + (size_t)(kt + 1) * BK);
      ra0 = pa[0]; ra1 = pa[1];
      const uint4* pb = (const uint4*)(gB + (size_t)(kt + 1) * BK);
      rb0 = pb[0]; rb1 = pb[1];
    }
#endif

    // --- fragment loads (ISA 16-bit A 16x32 / B 32x16 layouts) ---
    v16bf afrag[4], bfrag[2];
    const int koffA = lhi ? 8 : 0;    // lanes<16: K{0..7,16..23}; >=16: K{8..15,24..31}
    const int koffB = lhi ? 8 : 0;    // lanes<16: K{0..7,16..23}; >=16: K{8..15,24..31}
#pragma unroll
    for (int mi = 0; mi < 4; ++mi) {
      const __bf16* base = &As[cur][wm + mi * 16 + l15][0];
      uint4 c0 = *(const uint4*)(base + koffA);
      uint4 c1 = *(const uint4*)(base + koffA + 16);
      v8u u = {c0.x, c0.y, c0.z, c0.w, c1.x, c1.y, c1.z, c1.w};
      afrag[mi] = __builtin_bit_cast(v16bf, u);
    }
#pragma unroll
    for (int ni = 0; ni < 2; ++ni) {
      const __bf16* base = &Bs[cur][wn + ni * 16 + l15][0];
      uint4 c0 = *(const uint4*)(base + koffB);
      uint4 c1 = *(const uint4*)(base + koffB + 16);
      v8u u = {c0.x, c0.y, c0.z, c0.w, c1.x, c1.y, c1.z, c1.w};
      bfrag[ni] = __builtin_bit_cast(v16bf, u);
    }

#pragma unroll
    for (int mi = 0; mi < 4; ++mi)
#pragma unroll
      for (int ni = 0; ni < 2; ++ni)
        acc[mi][ni] = __builtin_amdgcn_wmma_f32_16x16x32_bf16(
            false, afrag[mi], false, bfrag[ni], (short)0, acc[mi][ni],
            false, false);

#if USE_ASYNC_LDS
    __syncthreads();    // all waves done reading cur before kt+2 overwrites it
#else
    if (kt + 1 < NK) {  // commit prefetched tile to the other buffer
      *(uint4*)&As[nxt][lrow][lcol]     = ra0;
      *(uint4*)&As[nxt][lrow][lcol + 8] = ra1;
      *(uint4*)&Bs[nxt][lrow][lcol]     = rb0;
      *(uint4*)&Bs[nxt][lrow][lcol + 8] = rb1;
    }
    __syncthreads();
#endif
  }

  // --- epilogue: C/D layout VGPR r -> M = r + 8*lane[4], N = lane[3:0] ---
#pragma unroll
  for (int ni = 0; ni < 2; ++ni) {
    const int col = bn + wn + ni * 16 + l15;
    const float bv = bias[col];
#pragma unroll
    for (int mi = 0; mi < 4; ++mi) {
#pragma unroll
      for (int r = 0; r < 8; ++r) {
        const int row = bm + wm + mi * 16 + lhi * 8 + r;
        out[(size_t)row * OUT_F + col] = acc[mi][ni][r] + bv;
      }
    }
  }
}

// ------------------------------- Launch ------------------------------------
extern "C" void kernel_launch(void* const* d_in, const int* in_sizes, int n_in,
                              void* d_out, int out_size, void* d_ws, size_t ws_size,
                              hipStream_t stream) {
  const float* x     = (const float*)d_in[0];
  const float* amp   = (const float*)d_in[1];
  const float* rf    = (const float*)d_in[2];
  const float* cf    = (const float*)d_in[3];
  const float* rp    = (const float*)d_in[4];
  const float* cp    = (const float*)d_in[5];
  const float* lA    = (const float*)d_in[6];
  const float* lB    = (const float*)d_in[7];
  const float* alpha = (const float*)d_in[8];
  const float* beta  = (const float*)d_in[9];
  const float* bias  = (const float*)d_in[10];
  float* out = (float*)d_out;

  // workspace layout (~96.5 MB): S | C | W_bf16 | x_bf16
  float*  S  = (float*)d_ws;
  float*  C  = S + (size_t)NFREQ * OUT_F;
  __bf16* Wb = (__bf16*)(C + (size_t)NFREQ * IN_F);
  __bf16* Xb = Wb + (size_t)OUT_F * IN_F;

  k_fourier_sc<<<dim3((NFREQ * 4096) / 256), 256, 0, stream>>>(amp, rf, cf, rp, cp, S, C);
  k_wsynth<<<dim3(IN_F / 1024, OUT_F), 256, 0, stream>>>(S, C, lA, lB, alpha, beta, Wb);
  k_xcvt<<<dim3(((size_t)NTOK * IN_F) / 2048), 256, 0, stream>>>(x, Xb);
  k_gemm<<<dim3(OUT_F / BN, NTOK / BM), 256, 0, stream>>>(Xb, Wb, bias, out);
}
